// StackedAttentionBlock_80556406604300
// MI455X (gfx1250) — matrix operand order repaired and checked by reading, not compiled, not run on hardware
//
#include <hip/hip_runtime.h>
#include <cmath>

// ---- CDNA5 WMMA types ----
typedef __attribute__((ext_vector_type(16))) __bf16       v16bf;
typedef __attribute__((ext_vector_type(2)))  __bf16       v2bf;
typedef __attribute__((ext_vector_type(8)))  float        v8f;
typedef __attribute__((ext_vector_type(2)))  float        v2f;
typedef __attribute__((ext_vector_type(8)))  unsigned int v8u;

union ABTile { v16bf bf; v8u u; };

// Single f32 -> bf16 (hardware cvt path).
__device__ __forceinline__ unsigned short f2bfbits(float f) {
    __bf16 h = (__bf16)f;
    unsigned short b;
    __builtin_memcpy(&b, &h, 2);
    return b;
}
// Pair conversion -> single v_cvt_pk_bf16_f32 producing the packed dword.
__device__ __forceinline__ unsigned pack2bf(float lo, float hi) {
    v2f f; f.x = lo; f.y = hi;
    v2bf b = __builtin_convertvector(f, v2bf);
    unsigned u;
    __builtin_memcpy(&u, &b, 4);
    return u;
}
// A-operand dword index within a 32-K bf16 chunk (16 dwords), per ISA 16-bit A layout
__device__ __forceinline__ int dA(int p, int hi) { return (p & 3) + 4 * hi + 8 * (p >> 2); }

__device__ __forceinline__ v8f wmma_bf16(v16bf a, v16bf b, v8f c) {
    // D = A(16x32) * B(32x16) + C, fp32 accumulate
    return __builtin_amdgcn_wmma_f32_16x16x32_bf16(false, a, false, b, (short)0, c, false, false);
}

__device__ __forceinline__ float redmax16(float v) {
#pragma unroll
    for (int m = 1; m < 16; m <<= 1) v = fmaxf(v, __shfl_xor(v, m, 32));
    return v;
}
__device__ __forceinline__ float redsum16(float v) {
#pragma unroll
    for (int m = 1; m < 16; m <<= 1) v += __shfl_xor(v, m, 32);
    return v;
}

constexpr int Bsz = 4, Cch = 128, Ntok = 4096, HD = 32, DFF = 512;

// Packed-weight tile table (each tile = 32 lanes x 8 dwords = 256 dwords = 1KB):
//   Wq: tiles [0,8)    (head*2 + j)          Wk: [8,16)      Wv: [16,24)
//   Wproj: [24,56)     (j*4 + kc)
//   W1:    [56,184)    (j*4 + kc)            W2: [184,312)   (j*16 + kc)
constexpr int TQ = 0, TK = 8, TV = 16, TP = 24, T1 = 56, T2 = 184, TTOT = 312;

// ============================================================================
// Kernel 0: pre-pack all weights as bf16 in WMMA B-operand layout.
// One dword per thread: tile*256 + lane*8 + p  <=  pack(W[k0][col], W[k0+1][col])
// with k0 = 32*kc + 2*(p&3) + 16*(p>>2) + 8*hi, col = 16*j + (lane&15).
// ============================================================================
__global__ void pack_weights_kernel(const float* __restrict__ Wq, const float* __restrict__ Wk,
                                    const float* __restrict__ Wv, const float* __restrict__ Wp,
                                    const float* __restrict__ W1, const float* __restrict__ W2,
                                    unsigned* __restrict__ wpk) {
    int flat = blockIdx.x * 256 + threadIdx.x;          // 312*256 total
    int tile = flat >> 8;
    int within = flat & 255;
    int lane = within >> 3, p = within & 7;
    int nlo = lane & 15, hi = lane >> 4;

    const float* src;
    int j, kc, cols;
    if (tile < TP) {                                    // Wq / Wk / Wv  [nh][32][32]
        int s = tile >> 3, rem = tile & 7;
        int h = rem >> 1;
        j = rem & 1; kc = 0; cols = HD;
        src = (s == 0 ? Wq : (s == 1 ? Wk : Wv)) + (size_t)h * HD * HD;
    } else if (tile < T1) {                             // Wproj [128][128]
        int t = tile - TP;
        j = t >> 2; kc = t & 3; cols = Cch; src = Wp;
    } else if (tile < T2) {                             // W1 [128][512]
        int t = tile - T1;
        j = t >> 2; kc = t & 3; cols = DFF; src = W1;
    } else {                                            // W2 [512][128]
        int t = tile - T2;
        j = t >> 4; kc = t & 15; cols = Cch; src = W2;
    }
    int k0  = 32 * kc + 2 * (p & 3) + 16 * (p >> 2) + 8 * hi;
    int col = 16 * j + nlo;
    wpk[flat] = pack2bf(src[(size_t)k0 * cols + col], src[(size_t)(k0 + 1) * cols + col]);
}

// Fetch one pre-packed B tile: a single contiguous 32-byte load per lane.
__device__ __forceinline__ void load_btile(ABTile& t, const unsigned* __restrict__ wpk,
                                           int tileIdx, int lane) {
    t.u = *(const v8u*)(wpk + (size_t)tileIdx * 256 + lane * 8);
}

// ============================================================================
// Kernel 1: per-head QKV projection.  xi = x[:,32h:32h+32,:] (+ prev acc)
// Writes Q,K row-major bf16 [B][N][32] and V transposed bf16 [B][32][N].
// ============================================================================
__global__ void qkv_kernel(const float* __restrict__ x, const float* __restrict__ outs,
                           const float* __restrict__ bq, const float* __restrict__ bk,
                           const float* __restrict__ bv, const unsigned* __restrict__ wpk,
                           unsigned short* __restrict__ Qb, unsigned short* __restrict__ Kb,
                           unsigned short* __restrict__ VT, int head) {
    const int lane = threadIdx.x & 31;
    const int wave = threadIdx.x >> 5;
    const int tile = blockIdx.x * 4 + wave;      // 1024 tiles of 16 tokens
    const int b    = tile >> 8;
    const int n0   = (tile & 255) << 4;
    const int nlo  = lane & 15, hi = lane >> 4;
    const int tok  = n0 + nlo;

    // A tile: 16 tokens x 32 channels (x slice + previous head accumulator)
    ABTile a;
#pragma unroll
    for (int p = 0; p < 8; ++p) {
        int kp = 2 * (p & 3) + 16 * (p >> 2) + 8 * hi;
        float v0 = x[((size_t)b * Cch + head * HD + kp) * Ntok + tok];
        float v1 = x[((size_t)b * Cch + head * HD + kp + 1) * Ntok + tok];
        if (head > 0) {
            size_t base = ((size_t)b * Ntok + tok) * Cch + (head - 1) * HD + kp;
            v0 += outs[base];
            v1 += outs[base + 1];
        }
        a.u[p] = pack2bf(v0, v1);
    }

    const float* bs[3] = {bq, bk, bv};
    const int    tb[3] = {TQ, TK, TV};
#pragma unroll
    for (int s = 0; s < 3; ++s) {
#pragma unroll
        for (int j = 0; j < 2; ++j) {
            int col = 16 * j + nlo;
            ABTile wb;
            load_btile(wb, wpk, tb[s] + head * 2 + j, lane);
            float bias = bs[s][head * HD + col];
            v8f c;
#pragma unroll
            for (int r = 0; r < 8; ++r) c[r] = bias;
            v8f d = wmma_bf16(a.bf, wb.bf, c);
            if (s < 2) {
                unsigned short* dst = (s == 0) ? Qb : Kb;
#pragma unroll
                for (int r = 0; r < 8; ++r) {
                    int mm = r + 8 * hi;
                    dst[((size_t)b * Ntok + n0 + mm) * HD + col] = f2bfbits(d[r]);
                }
            } else {
#pragma unroll
                for (int r = 0; r < 8; ++r) {
                    int mm = r + 8 * hi;
                    VT[((size_t)b * HD + col) * Ntok + n0 + mm] = f2bfbits(d[r]);
                }
            }
        }
    }
}

// ============================================================================
// Kernel 2: flash attention per head. One wave = 16 query rows, streams
// 32-key tiles: 2 score WMMAs + online softmax + 2 PV WMMAs per tile.
// Writes head output (fp32) into the concat buffer outs[:,:,32h:32h+32].
// ============================================================================
__global__ void attn_kernel(const unsigned short* __restrict__ Q,
                            const unsigned short* __restrict__ K,
                            const unsigned short* __restrict__ VT,
                            float* __restrict__ outs, int head) {
    __shared__ __align__(16) unsigned short plds[4][16][32]; // per-wave P bounce
    const int lane = threadIdx.x & 31;
    const int wave = threadIdx.x >> 5;
    const int tile = blockIdx.x * 4 + wave;
    const int b    = tile >> 8;
    const int n0   = (tile & 255) << 4;
    const int nlo  = lane & 15, hi = lane >> 4;

    const unsigned* Qu = (const unsigned*)Q;
    const unsigned* Ku = (const unsigned*)K;
    const unsigned* Vu = (const unsigned*)VT;
    unsigned*       pl = (unsigned*)&plds[wave][0][0];

    ABTile aQ;
#pragma unroll
    for (int p = 0; p < 8; ++p)
        aQ.u[p] = Qu[((size_t)b * Ntok + n0 + nlo) * 16 + dA(p, hi)];

    v8f acc0, acc1;
    float mrow[8], lrow[8];
#pragma unroll
    for (int r = 0; r < 8; ++r) { acc0[r] = 0.f; acc1[r] = 0.f; mrow[r] = -INFINITY; lrow[r] = 0.f; }

    const float qs = 0.17677669529663687f * 1.4426950408889634f; // (1/sqrt(32))*log2e

    for (int kb = 0; kb < Ntok; kb += 32) {
        // Prefetch next key/value tiles while this tile computes (global_prefetch_b8)
        if (kb + 32 < Ntok) {
            __builtin_prefetch(&Ku[((size_t)b * Ntok + kb + 32 + nlo) * 16], 0, 1);
            __builtin_prefetch(&Vu[((size_t)b * HD + nlo) * (Ntok / 2) + ((kb + 32) >> 1)], 0, 1);
            __builtin_prefetch(&Vu[((size_t)b * HD + 16 + nlo) * (Ntok / 2) + ((kb + 32) >> 1)], 0, 1);
        }
        ABTile bK0, bK1;
#pragma unroll
        for (int p = 0; p < 8; ++p) {
            int db = dA(p, hi);
            bK0.u[p] = Ku[((size_t)b * Ntok + kb + nlo) * 16 + db];
            bK1.u[p] = Ku[((size_t)b * Ntok + kb + 16 + nlo) * 16 + db];
        }
        v8f z;
#pragma unroll
        for (int r = 0; r < 8; ++r) z[r] = 0.f;
        v8f s0 = wmma_bf16(aQ.bf, bK0.bf, z);
        v8f s1 = wmma_bf16(aQ.bf, bK1.bf, z);

        float p0[8], p1[8];
#pragma unroll
        for (int r = 0; r < 8; ++r) {
            float t0 = s0[r] * qs, t1 = s1[r] * qs;
            float mx = redmax16(fmaxf(t0, t1));
            float mn = fmaxf(mrow[r], mx);
            float al = exp2f(mrow[r] - mn);
            mrow[r] = mn;
            p0[r] = exp2f(t0 - mn);
            p1[r] = exp2f(t1 - mn);
            lrow[r] = lrow[r] * al + redsum16(p0[r] + p1[r]);
            acc0[r] *= al;
            acc1[r] *= al;
        }
        // D-layout -> A-layout transpose of P via LDS (bf16)
#pragma unroll
        for (int r = 0; r < 8; ++r) {
            int mm = r + 8 * hi;
            plds[wave][mm][nlo]      = f2bfbits(p0[r]);
            plds[wave][mm][16 + nlo] = f2bfbits(p1[r]);
        }
        __syncthreads();
        ABTile aP, bV0, bV1;
#pragma unroll
        for (int p = 0; p < 8; ++p) {
            aP.u[p] = pl[nlo * 16 + dA(p, hi)];
            int dv = (kb >> 1) + dA(p, hi);
            bV0.u[p] = Vu[((size_t)b * HD + nlo) * (Ntok / 2) + dv];
            bV1.u[p] = Vu[((size_t)b * HD + 16 + nlo) * (Ntok / 2) + dv];
        }
        acc0 = wmma_bf16(aP.bf, bV0.bf, acc0);
        acc1 = wmma_bf16(aP.bf, bV1.bf, acc1);
        __syncthreads();
    }
#pragma unroll
    for (int r = 0; r < 8; ++r) {
        int mm = r + 8 * hi;
        float inv = 1.0f / lrow[r];
        size_t base = ((size_t)b * Ntok + n0 + mm) * Cch + head * HD;
        outs[base + nlo]      = acc0[r] * inv;
        outs[base + 16 + nlo] = acc1[r] * inv;
    }
}

// ============================================================================
// Kernel 3: x_attn = concat(outs) @ Wproj + bproj; xr = xt + x_attn; LN1.
// Writes xn1 as fp32 (for final residual) and bf16 (FFN operand).
// ============================================================================
__global__ void proj_ln_kernel(const float* __restrict__ outs, const float* __restrict__ x,
                               const unsigned* __restrict__ wpk, const float* __restrict__ bp,
                               const float* __restrict__ g1, const float* __restrict__ bn1,
                               float* __restrict__ xn1f, unsigned short* __restrict__ xn1b) {
    const int lane = threadIdx.x & 31;
    const int wave = threadIdx.x >> 5;
    const int tile = blockIdx.x * 4 + wave;
    const int b    = tile >> 8;
    const int n0   = (tile & 255) << 4;
    const int nlo  = lane & 15, hi = lane >> 4;

    ABTile aC[4];
#pragma unroll
    for (int kc = 0; kc < 4; ++kc)
#pragma unroll
        for (int p = 0; p < 8; ++p) {
            int kp = 32 * kc + 2 * (p & 3) + 16 * (p >> 2) + 8 * hi;
            const float* sp = outs + ((size_t)b * Ntok + n0 + nlo) * Cch + kp;
            aC[kc].u[p] = pack2bf(sp[0], sp[1]);
        }

    float vAll[8][8];
#pragma unroll
    for (int j = 0; j < 8; ++j) {
        int col = 16 * j + nlo;
        v8f d;
        float bias = bp[col];
#pragma unroll
        for (int r = 0; r < 8; ++r) d[r] = bias;
#pragma unroll
        for (int kc = 0; kc < 4; ++kc) {
            ABTile wb;
            load_btile(wb, wpk, TP + j * 4 + kc, lane);
            d = wmma_bf16(aC[kc].bf, wb.bf, d);
        }
#pragma unroll
        for (int r = 0; r < 8; ++r) {
            int mm = r + 8 * hi;
            vAll[j][r] = d[r] + x[((size_t)b * Cch + col) * Ntok + n0 + mm]; // residual
        }
    }
    // LayerNorm over 128 channels (rows live across 16 lanes x 8 col-tiles)
    float mean[8], rstd[8];
#pragma unroll
    for (int r = 0; r < 8; ++r) {
        float s = 0.f;
#pragma unroll
        for (int j = 0; j < 8; ++j) s += vAll[j][r];
        mean[r] = redsum16(s) * (1.0f / 128.0f);
    }
#pragma unroll
    for (int r = 0; r < 8; ++r) {
        float s = 0.f;
#pragma unroll
        for (int j = 0; j < 8; ++j) { float dd = vAll[j][r] - mean[r]; s += dd * dd; }
        rstd[r] = rsqrtf(redsum16(s) * (1.0f / 128.0f) + 1e-5f);
    }
#pragma unroll
    for (int j = 0; j < 8; ++j) {
        int col = 16 * j + nlo;
        float gg = g1[col], bb = bn1[col];
#pragma unroll
        for (int r = 0; r < 8; ++r) {
            int mm = r + 8 * hi;
            float y = (vAll[j][r] - mean[r]) * rstd[r] * gg + bb;
            size_t idx = ((size_t)b * Ntok + n0 + mm) * Cch + col;
            xn1f[idx] = y;
            xn1b[idx] = f2bfbits(y);
        }
    }
}

// ============================================================================
// Kernel 4: FFN h=gelu(xn1@W1+bf1) (bf16 staged in LDS), ff=h@W2+bf2,
// y=xn1+ff, LN2, transposed store to [B,C,H*W].
// ============================================================================
__global__ void ffn_kernel(const unsigned short* __restrict__ xn1b, const float* __restrict__ xn1f,
                           const unsigned* __restrict__ wpk, const float* __restrict__ bf1,
                           const float* __restrict__ bf2, const float* __restrict__ g2,
                           const float* __restrict__ bn2, float* __restrict__ out) {
    __shared__ __align__(16) unsigned short hsh[4][16][DFF]; // 64 KB / block
    const int lane = threadIdx.x & 31;
    const int wave = threadIdx.x >> 5;
    const int tile = blockIdx.x * 4 + wave;
    const int b    = tile >> 8;
    const int n0   = (tile & 255) << 4;
    const int nlo  = lane & 15, hi = lane >> 4;

    const unsigned* xu = (const unsigned*)xn1b;
    ABTile aX[4];
#pragma unroll
    for (int kc = 0; kc < 4; ++kc)
#pragma unroll
        for (int p = 0; p < 8; ++p)
            aX[kc].u[p] = xu[((size_t)b * Ntok + n0 + nlo) * 64 + kc * 16 + dA(p, hi)];

    // h = gelu(xn1 @ W1 + bf1): 32 col-tiles of 16, K=128 in 4 chunks
    for (int j = 0; j < 32; ++j) {
        int col = 16 * j + nlo;
        v8f d;
        float bias = bf1[col];
#pragma unroll
        for (int r = 0; r < 8; ++r) d[r] = bias;
#pragma unroll
        for (int kc = 0; kc < 4; ++kc) {
            ABTile wb;
            load_btile(wb, wpk, T1 + j * 4 + kc, lane);
            d = wmma_bf16(aX[kc].bf, wb.bf, d);
        }
#pragma unroll
        for (int r = 0; r < 8; ++r) {
            int mm = r + 8 * hi;
            float t = d[r];
            float gel = 0.5f * t * (1.0f + erff(t * 0.70710678118654752f)); // exact gelu
            hsh[wave][mm][col] = f2bfbits(gel);
        }
    }
    __syncthreads();

    const unsigned* hu = (const unsigned*)&hsh[wave][0][0];
    float vAll[8][8];
#pragma unroll
    for (int j2 = 0; j2 < 8; ++j2) {
        int col = 16 * j2 + nlo;
        v8f d;
        float bias = bf2[col];
#pragma unroll
        for (int r = 0; r < 8; ++r) d[r] = bias;
        for (int kc = 0; kc < 16; ++kc) {
            ABTile aH, wb;
#pragma unroll
            for (int p = 0; p < 8; ++p)
                aH.u[p] = hu[nlo * (DFF / 2) + kc * 16 + dA(p, hi)];
            load_btile(wb, wpk, T2 + j2 * 16 + kc, lane);
            d = wmma_bf16(aH.bf, wb.bf, d);
        }
#pragma unroll
        for (int r = 0; r < 8; ++r) {
            int mm = r + 8 * hi;
            vAll[j2][r] = d[r] + xn1f[((size_t)b * Ntok + n0 + mm) * Cch + col]; // residual
        }
    }
    // LN2
    float mean[8], rstd[8];
#pragma unroll
    for (int r = 0; r < 8; ++r) {
        float s = 0.f;
#pragma unroll
        for (int j = 0; j < 8; ++j) s += vAll[j][r];
        mean[r] = redsum16(s) * (1.0f / 128.0f);
    }
#pragma unroll
    for (int r = 0; r < 8; ++r) {
        float s = 0.f;
#pragma unroll
        for (int j = 0; j < 8; ++j) { float dd = vAll[j][r] - mean[r]; s += dd * dd; }
        rstd[r] = rsqrtf(redsum16(s) * (1.0f / 128.0f) + 1e-5f);
    }
#pragma unroll
    for (int j = 0; j < 8; ++j) {
        int col = 16 * j + nlo;
        float gg = g2[col], bb = bn2[col];
#pragma unroll
        for (int r = 0; r < 8; ++r) {
            int mm = r + 8 * hi;
            float y = (vAll[j][r] - mean[r]) * rstd[r] * gg + bb;
            out[((size_t)b * Cch + col) * Ntok + n0 + mm] = y; // [B,C,H,W] layout
        }
    }
}

// ============================================================================
extern "C" void kernel_launch(void* const* d_in, const int* in_sizes, int n_in,
                              void* d_out, int out_size, void* d_ws, size_t ws_size,
                              hipStream_t stream) {
    const float* x     = (const float*)d_in[0];
    const float* Wq    = (const float*)d_in[1];
    const float* bq    = (const float*)d_in[2];
    const float* Wk    = (const float*)d_in[3];
    const float* bk    = (const float*)d_in[4];
    const float* Wv    = (const float*)d_in[5];
    const float* bv    = (const float*)d_in[6];
    const float* Wproj = (const float*)d_in[7];
    const float* bproj = (const float*)d_in[8];
    const float* g1    = (const float*)d_in[9];
    const float* bn1   = (const float*)d_in[10];
    const float* W1    = (const float*)d_in[11];
    const float* bf1   = (const float*)d_in[12];
    const float* W2    = (const float*)d_in[13];
    const float* bf2   = (const float*)d_in[14];
    const float* g2    = (const float*)d_in[15];
    const float* bn2   = (const float*)d_in[16];
    float* out = (float*)d_out;

    // Workspace carve-up (bytes):
    char* w = (char*)d_ws;
    float*          outs = (float*)(w);                       //  8 MB: concat head outputs
    unsigned short* Qb   = (unsigned short*)(w + 8388608);    //  1 MB
    unsigned short* Kb   = (unsigned short*)(w + 9437184);    //  1 MB
    unsigned short* VT   = (unsigned short*)(w + 10485760);   //  1 MB
    float*          xn1f = (float*)(w + 11534336);            //  8 MB
    unsigned short* xn1b = (unsigned short*)(w + 19922944);   //  4 MB
    unsigned*       wpk  = (unsigned*)(w + 24117248);         // 312 KB packed weights

    dim3 block(128); // 4 waves/block, 1 wave = 16-token tile
    dim3 grid(256);

    pack_weights_kernel<<<dim3(TTOT), dim3(256), 0, stream>>>(Wq, Wk, Wv, Wproj, W1, W2, wpk);

    for (int h = 0; h < 4; ++h) { // heads chained through the accumulator
        qkv_kernel<<<grid, block, 0, stream>>>(x, outs, bq, bk, bv, wpk, Qb, Kb, VT, h);
        attn_kernel<<<grid, block, 0, stream>>>(Qb, Kb, VT, outs, h);
    }
    proj_ln_kernel<<<grid, block, 0, stream>>>(outs, x, wpk, bproj, g1, bn1, xn1f, xn1b);
    ffn_kernel<<<grid, block, 0, stream>>>(xn1b, xn1f, wpk, bf1, bf2, g2, bn2, out);
}
